// LLaDABlock_1211180777699
// MI455X (gfx1250) — matrix operand order repaired and checked by reading, not compiled, not run on hardware
//
#include <hip/hip_runtime.h>
#include <hip/hip_bf16.h>
#include <cstdint>

// ---------------------------------------------------------------------------
// Types for CDNA5 WMMA (wave32): v_wmma_f32_16x16x32_bf16
// ---------------------------------------------------------------------------
typedef __attribute__((ext_vector_type(16))) __bf16 v16bf;
typedef __attribute__((ext_vector_type(8)))  __bf16 v8bf;   // 16 bytes
typedef __attribute__((ext_vector_type(8)))  float  v8f;
typedef __attribute__((ext_vector_type(4)))  unsigned int v4u;
typedef __attribute__((ext_vector_type(8)))  int    v8i;
typedef __attribute__((ext_vector_type(4)))  int    v4i;

union V16U { v16bf v; v8bf h[2]; };

__device__ __forceinline__ v8f wmma_bf16(v16bf a, v16bf b, v8f c) {
    return __builtin_amdgcn_wmma_f32_16x16x32_bf16(false, a, false, b,
                                                   (short)0, c, false, false);
}

// bf16 <-> f32 (round-to-nearest-even on pack)
__device__ __forceinline__ unsigned short f2bf(float f) {
    union { float f; unsigned u; } x; x.f = f;
    unsigned u = x.u;
    unsigned r = (u + 0x7FFFu + ((u >> 16) & 1u)) >> 16;
    return (unsigned short)r;
}
__device__ __forceinline__ float bf2f(unsigned short h) {
    union { unsigned u; float f; } x; x.u = ((unsigned)h) << 16;
    return x.f;
}

// A fragment (16x32 bf16, MxK): lane l: m = l&15, hi = l>>4.
__device__ __forceinline__ v16bf load_afrag(const unsigned short* base, int stride,
                                            int m, int k0, int hi) {
    V16U u;
    const unsigned short* p = base + (size_t)m * stride + k0 + hi * 8;
    u.h[0] = *reinterpret_cast<const v8bf*>(p);
    u.h[1] = *reinterpret_cast<const v8bf*>(p + 16);
    return u.v;
}
// B fragment (32x16 bf16, KxN) from K-contiguous (B-transposed) tile.
__device__ __forceinline__ v16bf load_bfrag(const unsigned short* base, int stride,
                                            int n, int k0, int hi) {
    V16U u;
    const unsigned short* p = base + (size_t)n * stride + k0 + hi * 8;
    u.h[0] = *reinterpret_cast<const v8bf*>(p);
    u.h[1] = *reinterpret_cast<const v8bf*>(p + 16);
    return u.v;
}

// ---------------------------------------------------------------------------
// Tensor Data Mover: 2D tile (bf16) global -> LDS with row padding.
// D# layout per CDNA5 ISA:
//   g0: [1:0]=count=1, [63:32]=lds_addr, [120:64]=global_addr, [127:126]=type=2
//   g1: data_size=1(2B), pad_enable, pad_interval (2^(v+1) DWORDs),
//       pad_amount (v+1 DWORDs), tensor_dim0/1, tile_dim0/1, dim0_stride
// Groups 2/3 zero (2D tensor); 6-arg builtin form on this toolchain.
// ---------------------------------------------------------------------------
__device__ __forceinline__ void tdm_load_2d_bf16(const void* gaddr, unsigned lds_off,
                                                 unsigned tensor_d0, unsigned tensor_d1,
                                                 unsigned tile_d0, unsigned tile_d1,
                                                 unsigned long long stride0_elems,
                                                 unsigned pad_interval, unsigned pad_amount) {
    unsigned long long ga = (unsigned long long)(uintptr_t)gaddr;
    v4u g0 = { 1u,                                   // count=1, user mode
               lds_off,                              // lds_addr (bytes)
               (unsigned)(ga & 0xFFFFFFFFu),
               ((unsigned)((ga >> 32) & 0x01FFFFFFu)) | 0x80000000u }; // type=2
    unsigned w0 = (1u << 16)                 // data_size = 1 (2 bytes)
                | (1u << 20)                 // pad_enable
                | (pad_interval << 22)
                | (pad_amount << 25);
    unsigned w1 = (tensor_d0 & 0xFFFFu) << 16;                       // dim0 lo16 -> [63:48]
    unsigned w2 = (tensor_d0 >> 16) | ((tensor_d1 & 0xFFFFu) << 16); // dim0 hi / dim1 lo
    unsigned w3 = (tensor_d1 >> 16) | (tile_d0 << 16);               // dim1 hi / tile_dim0
    unsigned w4 = tile_d1 & 0xFFFFu;                                 // tile_dim1 (tile_dim2=0)
    unsigned w5 = (unsigned)(stride0_elems & 0xFFFFFFFFull);         // dim0_stride lo32
    unsigned w6 = (unsigned)((stride0_elems >> 32) & 0xFFFFull);     // dim0_stride hi16
    v8i g1 = { (int)w0, (int)w1, (int)w2, (int)w3, (int)w4, (int)w5, (int)w6, 0 };
    v4i gz4 = { 0, 0, 0, 0 };
    v8i gz8 = { 0, 0, 0, 0, 0, 0, 0, 0 };
    __builtin_amdgcn_tensor_load_to_lds(g0, g1, gz4, gz4, gz8, 0);
}

// ---------------------------------------------------------------------------
// Model constants
// ---------------------------------------------------------------------------
#define TT   2048
#define CC   2048
#define HH   16
#define DD   128
#define BSZ  128
#define NKB  16          // key blocks
#define TOPK 4
#define HID  8192
#define LPAD 136         // padded LDS row (136*2 = 272B, 16B aligned, bank-rotating)

// ---------------------------------------------------------------------------
// fp32 -> bf16 weight conversion
// ---------------------------------------------------------------------------
__global__ void f32_to_bf16_kernel(const float* __restrict__ in,
                                   unsigned short* __restrict__ out, size_t n) {
    size_t i = (size_t)blockIdx.x * blockDim.x + threadIdx.x;
    size_t stride = (size_t)gridDim.x * blockDim.x;
    for (; i < n; i += stride) out[i] = f2bf(in[i]);
}

// ---------------------------------------------------------------------------
// LayerNorm (weight only, no bias) -> bf16
// ---------------------------------------------------------------------------
__global__ __launch_bounds__(256)
void ln_bf16_kernel(const float* __restrict__ x, const float* __restrict__ w,
                    unsigned short* __restrict__ out) {
    __shared__ float red[256];
    const int row = blockIdx.x, tid = threadIdx.x;
    const float* xr = x + (size_t)row * CC;
    float s = 0.f;
    for (int i = tid; i < CC; i += 256) s += xr[i];
    red[tid] = s; __syncthreads();
    for (int off = 128; off > 0; off >>= 1) {
        if (tid < off) red[tid] += red[tid + off];
        __syncthreads();
    }
    const float mean = red[0] * (1.0f / CC);
    __syncthreads();
    float v = 0.f;
    for (int i = tid; i < CC; i += 256) { float d = xr[i] - mean; v += d * d; }
    red[tid] = v; __syncthreads();
    for (int off = 128; off > 0; off >>= 1) {
        if (tid < off) red[tid] += red[tid + off];
        __syncthreads();
    }
    const float rstd = rsqrtf(red[0] * (1.0f / CC) + 1e-5f);
    __syncthreads();
    for (int i = tid; i < CC; i += 256)
        out[(size_t)row * CC + i] = f2bf((xr[i] - mean) * rstd * w[i]);
}

// ---------------------------------------------------------------------------
// Block-sparse attention: one workgroup per (q-block, head). 8 waves.
// K tiles arrive via the Tensor Data Mover (TENSORcnt); V^T/P via vector path.
// ---------------------------------------------------------------------------
__global__ __launch_bounds__(256)
void blocksparse_attn_kernel(const unsigned short* __restrict__ xn,
                             unsigned short* __restrict__ att) {
    extern __shared__ char smem[];
    unsigned short* kv = (unsigned short*)smem;          // 128 x LPAD bf16
    unsigned short* ps = kv + 128 * LPAD;                // 128 x LPAD bf16
    float* lse_b   = (float*)(ps + 128 * LPAD);          // [128][NKB]
    float* lse_sel = lse_b + 128 * NKB;                  // [128]
    float* bscore  = lse_sel + 128;                      // [NKB]
    int*   sel     = (int*)(bscore + NKB);               // [TOPK]

    const int qb = blockIdx.x, h = blockIdx.y;
    const int tid = threadIdx.x;
    const int lane = tid & 31, wave = tid >> 5;
    const int ln = lane & 15, hi = lane >> 4;
    const int q0 = qb * BSZ;
    const int cb = h * DD;
    const int wm = wave * 16;
    const float scale = 0.08838834764831845f; // 1/sqrt(128)
    const unsigned kv_off = (unsigned)(uintptr_t)(void*)kv;

    // Q fragments in registers: 4 k-steps of 32 over head dim 128
    v16bf qf[4];
#pragma unroll
    for (int ks = 0; ks < 4; ++ks)
        qf[ks] = load_afrag(xn, CC, q0 + wm + ln, cb + ks * 32, hi);

    const v8f vzero = {0.f, 0.f, 0.f, 0.f, 0.f, 0.f, 0.f, 0.f};

    // ---------------- Pass 1: LSE of every 128-key block --------------------
    for (int kb = 0; kb < NKB; ++kb) {
        __syncthreads();
        // K tile via TDM: 128 rows x 128 cols bf16, row stride CC, pad 128->136
        // row = 256B = 64 DWORDs -> pad_interval=5 ; pad 16B = 4 DWORDs -> 3
        if (wave == 0) {
            tdm_load_2d_bf16(xn + (size_t)(kb * BSZ) * CC + cb, kv_off,
                             CC, TT, 128, 128, (unsigned long long)CC, 5u, 3u);
            __builtin_amdgcn_s_wait_tensorcnt((short)0);
        }
        __syncthreads();

        v8f acc[8];
#pragma unroll
        for (int j = 0; j < 8; ++j) acc[j] = vzero;
#pragma unroll
        for (int ks = 0; ks < 4; ++ks) {
            v16bf a = qf[ks];
#pragma unroll
            for (int j = 0; j < 8; ++j) {
                v16bf b = load_bfrag(kv, LPAD, j * 16 + ln, ks * 32, hi);
                acc[j] = wmma_bf16(a, b, acc[j]);
            }
        }
#pragma unroll
        for (int r = 0; r < 8; ++r) {
            float mx = -3.0e38f;
#pragma unroll
            for (int j = 0; j < 8; ++j) mx = fmaxf(mx, acc[j][r] * scale);
            mx = fmaxf(mx, __shfl_xor(mx, 1, 32));
            mx = fmaxf(mx, __shfl_xor(mx, 2, 32));
            mx = fmaxf(mx, __shfl_xor(mx, 4, 32));
            mx = fmaxf(mx, __shfl_xor(mx, 8, 32));
            float s = 0.f;
#pragma unroll
            for (int j = 0; j < 8; ++j) s += __expf(acc[j][r] * scale - mx);
            s += __shfl_xor(s, 1, 32);
            s += __shfl_xor(s, 2, 32);
            s += __shfl_xor(s, 4, 32);
            s += __shfl_xor(s, 8, 32);
            if (ln == 0)
                lse_b[(wm + r + 8 * hi) * NKB + kb] = mx + __logf(s);
        }
    }

    // ---------------- Block scores + top-4 ---------------------------------
    __syncthreads();
    if (tid < NKB) bscore[tid] = 0.f;
    __syncthreads();
    if (tid < 128) {
        float m = -3.0e38f;
        for (int kb = 0; kb < NKB; ++kb) m = fmaxf(m, lse_b[tid * NKB + kb]);
        float t = 0.f;
        for (int kb = 0; kb < NKB; ++kb) t += __expf(lse_b[tid * NKB + kb] - m);
        float tot = m + __logf(t);
        for (int kb = 0; kb < NKB; ++kb)
            atomicAdd(&bscore[kb], __expf(lse_b[tid * NKB + kb] - tot));
    }
    __syncthreads();
    if (tid == 0) {
        unsigned taken = 0;
        for (int t = 0; t < TOPK; ++t) {
            int best = 0; float bv = -3.0e38f;
            for (int kb = 0; kb < NKB; ++kb)
                if (!((taken >> kb) & 1u) && bscore[kb] > bv) { bv = bscore[kb]; best = kb; }
            sel[t] = best; taken |= 1u << best;
        }
    }
    __syncthreads();
    if (tid < 128) {
        float m = -3.0e38f;
        for (int t = 0; t < TOPK; ++t) m = fmaxf(m, lse_b[tid * NKB + sel[t]]);
        float s = 0.f;
        for (int t = 0; t < TOPK; ++t) s += __expf(lse_b[tid * NKB + sel[t]] - m);
        lse_sel[tid] = m + __logf(s);
    }
    __syncthreads();

    float lsev[8];
#pragma unroll
    for (int r = 0; r < 8; ++r) lsev[r] = lse_sel[wm + r + 8 * hi];

    // ---------------- Pass 2: masked attention over kept blocks ------------
    v8f o[8];
#pragma unroll
    for (int j = 0; j < 8; ++j) o[j] = vzero;

    for (int b = 0; b < TOPK; ++b) {
        const int kbi = sel[b];
        __syncthreads();
        if (wave == 0) {
            tdm_load_2d_bf16(xn + (size_t)(kbi * BSZ) * CC + cb, kv_off,
                             CC, TT, 128, 128, (unsigned long long)CC, 5u, 3u);
            __builtin_amdgcn_s_wait_tensorcnt((short)0);
        }
        __syncthreads();

        v8f acc[8];
#pragma unroll
        for (int j = 0; j < 8; ++j) acc[j] = vzero;
#pragma unroll
        for (int ks = 0; ks < 4; ++ks) {
            v16bf a = qf[ks];
#pragma unroll
            for (int j = 0; j < 8; ++j) {
                v16bf bfr = load_bfrag(kv, LPAD, j * 16 + ln, ks * 32, hi);
                acc[j] = wmma_bf16(a, bfr, acc[j]);
            }
        }
        // P = exp(logit - lse_sel), staged to LDS in A-layout rows
#pragma unroll
        for (int j = 0; j < 8; ++j)
#pragma unroll
            for (int r = 0; r < 8; ++r) {
                float p = __expf(acc[j][r] * scale - lsev[r]);
                ps[(size_t)(wm + r + 8 * hi) * LPAD + j * 16 + ln] = f2bf(p);
            }
        __syncthreads();
        // V^T tile (Vt[d][key]) overwrites kv (vector path: TDM cannot transpose)
        for (int i = 0; i < 8; ++i) {
            int c = i * 256 + tid;
            int key = c >> 4, dcol = (c & 15) * 8;
            v8bf vv = *reinterpret_cast<const v8bf*>(
                xn + (size_t)(kbi * BSZ + key) * CC + cb + dcol);
            const unsigned short* vs = reinterpret_cast<const unsigned short*>(&vv);
#pragma unroll
            for (int e = 0; e < 8; ++e)
                kv[(size_t)(dcol + e) * LPAD + key] = vs[e];
        }
        __syncthreads();
#pragma unroll
        for (int ks = 0; ks < 4; ++ks) {
            v16bf a = load_afrag(ps, LPAD, wm + ln, ks * 32, hi);
#pragma unroll
            for (int j = 0; j < 8; ++j) {
                v16bf bfr = load_bfrag(kv, LPAD, j * 16 + ln, ks * 32, hi);
                o[j] = wmma_bf16(a, bfr, o[j]);
            }
        }
    }

#pragma unroll
    for (int j = 0; j < 8; ++j)
#pragma unroll
        for (int r = 0; r < 8; ++r)
            att[(size_t)(q0 + wm + r + 8 * hi) * CC + cb + j * 16 + ln] = f2bf(o[j][r]);
}

// ---------------------------------------------------------------------------
// Generic bf16 WMMA GEMM: C[M,N] = A[M,K] @ B[K,N]
// A tile via TDM (padded rows); B tile via vector path (needs transpose).
// mode 0: out = f32, += residual(f32). mode 1: out = bf16.
// ---------------------------------------------------------------------------
__global__ __launch_bounds__(256)
void gemm_bf16_kernel(const unsigned short* __restrict__ A,
                      const unsigned short* __restrict__ B,
                      const float* __restrict__ residual,
                      void* __restrict__ out,
                      int M, int N, int K, int mode) {
    __shared__ unsigned short As[128 * 72];   // [m][k], padded 64->72
    __shared__ unsigned short Bs[128 * 72];   // [n][k], padded (B transposed)

    const int n0 = blockIdx.x * 128, m0 = blockIdx.y * 128;
    const int tid = threadIdx.x;
    const int lane = tid & 31, wave = tid >> 5;
    const int ln = lane & 15, hi = lane >> 4;
    const int wr = wave >> 1, wc = wave & 1;   // 4x2 wave grid: 32 rows x 64 cols
    const unsigned As_off = (unsigned)(uintptr_t)(void*)As;

    const v8f vzero = {0.f, 0.f, 0.f, 0.f, 0.f, 0.f, 0.f, 0.f};
    v8f acc[2][4];
#pragma unroll
    for (int mt = 0; mt < 2; ++mt)
#pragma unroll
        for (int j = 0; j < 4; ++j) acc[mt][j] = vzero;

    for (int k0 = 0; k0 < K; k0 += 64) {
        __syncthreads();
        // A tile 128x64 via TDM: row = 128B = 32 DWORDs -> pad_interval=4;
        // pad 16B = 4 DWORDs -> pad_amount=3 (64 -> 72 elements)
        if (wave == 0)
            tdm_load_2d_bf16(A + (size_t)m0 * K + k0, As_off,
                             (unsigned)K, (unsigned)M, 64, 128,
                             (unsigned long long)K, 4u, 3u);
        // B tile 64x128, transposed into Bs[n][k] (vector path)
#pragma unroll
        for (int i = 0; i < 4; ++i) {
            int c = i * 256 + tid;
            int kr = c >> 4, nc = (c & 15) * 8;
            const unsigned short* gp = B + (size_t)(k0 + kr) * N + n0 + nc;
            v8bf bb = *reinterpret_cast<const v8bf*>(gp);
            __builtin_prefetch(gp + (size_t)64 * N, 0, 0);
            const unsigned short* bs = reinterpret_cast<const unsigned short*>(&bb);
#pragma unroll
            for (int e = 0; e < 8; ++e) Bs[(nc + e) * 72 + kr] = bs[e];
        }
        if (wave == 0) __builtin_amdgcn_s_wait_tensorcnt((short)0);
        __syncthreads();
#pragma unroll
        for (int ks = 0; ks < 64; ks += 32) {
            v16bf a[2];
#pragma unroll
            for (int mt = 0; mt < 2; ++mt)
                a[mt] = load_afrag(As, 72, wr * 32 + mt * 16 + ln, ks, hi);
#pragma unroll
            for (int j = 0; j < 4; ++j) {
                v16bf b = load_bfrag(Bs, 72, wc * 64 + j * 16 + ln, ks, hi);
#pragma unroll
                for (int mt = 0; mt < 2; ++mt)
                    acc[mt][j] = wmma_bf16(a[mt], b, acc[mt][j]);
            }
        }
    }

#pragma unroll
    for (int mt = 0; mt < 2; ++mt)
#pragma unroll
        for (int j = 0; j < 4; ++j)
#pragma unroll
            for (int r = 0; r < 8; ++r) {
                int m = m0 + wr * 32 + mt * 16 + r + 8 * hi;
                int n = n0 + wc * 64 + j * 16 + ln;
                float v = acc[mt][j][r];
                if (mode == 0) {
                    ((float*)out)[(size_t)m * N + n] =
                        v + residual[(size_t)m * N + n];
                } else {
                    ((unsigned short*)out)[(size_t)m * N + n] = f2bf(v);
                }
            }
}

// ---------------------------------------------------------------------------
// SwiGLU gate: h = silu(g) * u   (bf16 in, bf16 out)
// ---------------------------------------------------------------------------
__global__ void silu_mul_kernel(const unsigned short* __restrict__ g,
                                const unsigned short* __restrict__ u,
                                unsigned short* __restrict__ out, size_t n) {
    size_t i = (size_t)blockIdx.x * blockDim.x + threadIdx.x;
    size_t stride = (size_t)gridDim.x * blockDim.x;
    for (; i < n; i += stride) {
        float a = bf2f(g[i]);
        float s = a / (1.f + __expf(-a));
        out[i] = f2bf(s * bf2f(u[i]));
    }
}

// ---------------------------------------------------------------------------
// Host orchestration
// ---------------------------------------------------------------------------
extern "C" void kernel_launch(void* const* d_in, const int* in_sizes, int n_in,
                              void* d_out, int out_size, void* d_ws, size_t ws_size,
                              hipStream_t stream) {
    (void)in_sizes; (void)n_in; (void)out_size; (void)ws_size;

    const float* x          = (const float*)d_in[0];
    const float* attn_w     = (const float*)d_in[1];
    const float* ff_w       = (const float*)d_in[2];
    const float* w_attn_out = (const float*)d_in[3];
    const float* w_ff_proj  = (const float*)d_in[4];
    const float* w_up_proj  = (const float*)d_in[5];
    const float* w_ff_out   = (const float*)d_in[6];
    float* out = (float*)d_out;

    char* ws = (char*)d_ws;
    auto carve = [&](size_t bytes) {
        char* p = ws;
        ws += (bytes + 255) & ~(size_t)255;
        return p;
    };
    unsigned short* xn     = (unsigned short*)carve((size_t)TT * CC * 2);
    unsigned short* att    = (unsigned short*)carve((size_t)TT * CC * 2);
    float*          x2     = (float*)         carve((size_t)TT * CC * 4);
    unsigned short* xn2    = (unsigned short*)carve((size_t)TT * CC * 2);
    unsigned short* h1     = (unsigned short*)carve((size_t)TT * HID * 2);
    unsigned short* h2     = (unsigned short*)carve((size_t)TT * HID * 2);
    unsigned short* hh     = (unsigned short*)carve((size_t)TT * HID * 2);
    unsigned short* wb_ao  = (unsigned short*)carve((size_t)CC * CC * 2);
    unsigned short* wb_fp  = (unsigned short*)carve((size_t)CC * HID * 2);
    unsigned short* wb_up  = (unsigned short*)carve((size_t)CC * HID * 2);
    unsigned short* wb_fo  = (unsigned short*)carve((size_t)HID * CC * 2);

    f32_to_bf16_kernel<<<2048, 256, 0, stream>>>(w_attn_out, wb_ao, (size_t)CC * CC);
    f32_to_bf16_kernel<<<4096, 256, 0, stream>>>(w_ff_proj, wb_fp, (size_t)CC * HID);
    f32_to_bf16_kernel<<<4096, 256, 0, stream>>>(w_up_proj, wb_up, (size_t)CC * HID);
    f32_to_bf16_kernel<<<4096, 256, 0, stream>>>(w_ff_out, wb_fo, (size_t)HID * CC);

    ln_bf16_kernel<<<TT, 256, 0, stream>>>(x, attn_w, xn);

    size_t attn_lds = (size_t)(2 * 128 * LPAD) * 2
                    + (size_t)(128 * NKB + 128 + NKB) * 4
                    + (size_t)TOPK * 4;
    blocksparse_attn_kernel<<<dim3(NKB, HH), 256, attn_lds, stream>>>(xn, att);

    gemm_bf16_kernel<<<dim3(CC / 128, TT / 128), 256, 0, stream>>>(
        att, wb_ao, x, (void*)x2, TT, CC, CC, 0);

    ln_bf16_kernel<<<TT, 256, 0, stream>>>(x2, ff_w, xn2);

    gemm_bf16_kernel<<<dim3(HID / 128, TT / 128), 256, 0, stream>>>(
        xn2, wb_fp, nullptr, (void*)h1, TT, HID, CC, 1);
    gemm_bf16_kernel<<<dim3(HID / 128, TT / 128), 256, 0, stream>>>(
        xn2, wb_up, nullptr, (void*)h2, TT, HID, CC, 1);
    silu_mul_kernel<<<4096, 256, 0, stream>>>(h1, h2, hh, (size_t)TT * HID);

    gemm_bf16_kernel<<<dim3(CC / 128, TT / 128), 256, 0, stream>>>(
        hh, wb_fo, x2, (void*)out, TT, CC, HID, 0);
}
